// Encoder_84069689852383
// MI455X (gfx1250) — matrix operand order repaired and checked by reading, not compiled, not run on hardware
//
#include <hip/hip_runtime.h>

// GRU encoder: V=100000, D=256, U=512, B=64, S=128, 3U=1536. f32 I/O,
// f16 WMMA (f32 accumulate) internally, async-to-LDS staging (CDNA5).

#define D_   256
#define U_   512
#define B_   64
#define S_   128
#define N3U  1536
#define LDH  40   // LDS row stride in halves: 80B rows (16B aligned), u32 stride 20

typedef __attribute__((ext_vector_type(16))) _Float16 v16h;
typedef __attribute__((ext_vector_type(8)))  float    v8f;

union FragU { unsigned int u[8]; v16h h; };
struct __align__(8) H4 { _Float16 v[4]; };

// ---- CDNA5 async global->LDS copies (ASYNCcnt) -----------------------------
__device__ __forceinline__ unsigned lds_off(const void* p) {
  // flat shared address: low 32 bits carry the LDS byte address
  return (unsigned)(unsigned long long)p;
}
__device__ __forceinline__ void async_b128(unsigned l, const void* g) {
  asm volatile("global_load_async_to_lds_b128 %0, %1, off"
               :: "v"(l), "v"(g) : "memory");
}
__device__ __forceinline__ void async_b64(unsigned l, const void* g) {
  asm volatile("global_load_async_to_lds_b64 %0, %1, off"
               :: "v"(l), "v"(g) : "memory");
}
__device__ __forceinline__ void wait_async_le4() {
  asm volatile("s_wait_asynccnt 0x4" ::: "memory");
}
__device__ __forceinline__ void wait_async0() {
  asm volatile("s_wait_asynccnt 0x0" ::: "memory");
}

// ---- WMMA fragment loads (ISA 7.12.2 layouts) ------------------------------
// A 16x32 f16: lane L row L&15; V0..V3=K(kh*8+2v), V4..V7=K(16+kh*8+2v), kh=L>>4
__device__ __forceinline__ v16h load_a_frag(const _Float16* rowptr, int lane) {
  const unsigned int* p = (const unsigned int*)rowptr;
  const int kh = (lane >> 4) & 1;
  FragU f;
#pragma unroll
  for (int v = 0; v < 4; ++v) f.u[v] = p[kh * 4 + v];
#pragma unroll
  for (int v = 0; v < 4; ++v) f.u[4 + v] = p[8 + kh * 4 + v];
  return f.h;
}
// B 32x16 f16 (LDS stored transposed [n][k]): lane col L&15;
// V0..V3=K(kh*8+2v), V4..V7=K(16+kh*8+2v), kh=L>>4 (same as A)
__device__ __forceinline__ v16h load_b_frag(const _Float16* colptr, int lane) {
  const unsigned int* p = (const unsigned int*)colptr;
  const int kh = (lane >> 4) & 1;
  FragU f;
#pragma unroll
  for (int v = 0; v < 4; ++v) f.u[v] = p[kh * 4 + v];
#pragma unroll
  for (int v = 0; v < 4; ++v) f.u[4 + v] = p[8 + kh * 4 + v];
  return f.h;
}

__device__ __forceinline__ float sigmoid_f(float x) {
  return 1.0f / (1.0f + __expf(-x));
}

// ---------------------------------------------------------------------------
// Setup: transpose-convert f32 [K][N] -> f16 [N][K]  (for kernel & Rk)
// ---------------------------------------------------------------------------
__global__ __launch_bounds__(256)
void transpose_cvt(const float* __restrict__ src, _Float16* __restrict__ dst,
                   int K, int N) {
  __shared__ _Float16 tile[32][33];
  const int bn = blockIdx.x * 32, bk = blockIdx.y * 32;
  const int t = threadIdx.x;
  {
    int nl = t & 31, kl0 = t >> 5;
#pragma unroll
    for (int j = 0; j < 4; ++j) {
      int kl = kl0 + j * 8;
      tile[kl][nl] = (_Float16)src[(bk + kl) * N + bn + nl];
    }
  }
  __syncthreads();
  {
    int kl = t & 31, nl0 = t >> 5;
#pragma unroll
    for (int j = 0; j < 4; ++j) {
      int nl = nl0 + j * 8;
      dst[(bn + nl) * K + bk + kl] = tile[kl][nl];
    }
  }
}

__global__ __launch_bounds__(256)
void cvt_f32_to_f16(const float* __restrict__ src, _Float16* __restrict__ dst,
                    int n) {
  int i = blockIdx.x * blockDim.x + threadIdx.x;
  if (i < n) dst[i] = (_Float16)src[i];
}

// ---------------------------------------------------------------------------
// Phase 1: xp = embedding[x] @ kernel + b_in. M=8192, N=1536, K=256.
// Tile 128x64, 8 waves. B async-copied from pre-converted f16 kernT[n][k];
// A gathered f32 -> f16 with float4 loads + packed 8B LDS stores.
// ---------------------------------------------------------------------------
__global__ __launch_bounds__(256)
void gru_input_proj(const int* __restrict__ x,
                    const float* __restrict__ emb,
                    const _Float16* __restrict__ kernT,   // [N3U][D_] f16
                    const float* __restrict__ b_in,
                    float* __restrict__ xp) {
  __shared__ int s_tok[128];
  __shared__ __align__(16) _Float16 sA[128 * LDH];
  __shared__ __align__(16) _Float16 sB[64 * LDH];

  const int t = threadIdx.x, lane = t & 31, w = t >> 5;
  const int mtile = blockIdx.x;   // 64 tiles of 128 rows
  const int ntile = blockIdx.y;   // 24 tiles of 64 cols

  if (t < 128) s_tok[t] = x[mtile * 128 + t];
  __syncthreads();

  v8f acc[4] = {};

  for (int kk = 0; kk < D_; kk += 32) {
    // B tile: 64 rows x 64B, async b64 x8/row = 512 xfers -> 2/thread (uniform)
#pragma unroll
    for (int i = 0; i < 2; ++i) {
      int e = t + i * 256;
      int rr = e >> 3, q = e & 7;
      async_b64(lds_off(&sB[rr * LDH + q * 4]),
                kernT + (ntile * 64 + rr) * D_ + kk + q * 4);
    }
    // A tile: 128 gathered rows x 32 f32 -> f16; 1024 float4 loads -> 4/thread
#pragma unroll
    for (int i = 0; i < 4; ++i) {
      int e = t + i * 256;
      int rr = e >> 3, k4 = (e & 7) * 4;
      const float4 f = *reinterpret_cast<const float4*>(
          &emb[s_tok[rr] * D_ + kk + k4]);
      H4 h;
      h.v[0] = (_Float16)f.x; h.v[1] = (_Float16)f.y;
      h.v[2] = (_Float16)f.z; h.v[3] = (_Float16)f.w;
      *reinterpret_cast<H4*>(&sA[rr * LDH + k4]) = h;
    }
    wait_async0();
    __syncthreads();

    v16h a = load_a_frag(&sA[(w * 16 + (lane & 15)) * LDH], lane);
#pragma unroll
    for (int c = 0; c < 4; ++c) {
      v16h b = load_b_frag(&sB[(c * 16 + (lane & 15)) * LDH], lane);
      acc[c] = __builtin_amdgcn_wmma_f32_16x16x32_f16(
          false, a, false, b, (short)0, acc[c], false, false);
    }
    __syncthreads();
  }

#pragma unroll
  for (int c = 0; c < 4; ++c) {
    int n = ntile * 64 + c * 16 + (lane & 15);
    float bi = b_in[n];
#pragma unroll
    for (int v = 0; v < 8; ++v) {
      int m = mtile * 128 + w * 16 + v + ((lane >> 4) << 3);
      xp[(long long)m * N3U + n] = acc[c][v] + bi;
    }
  }
}

// ---------------------------------------------------------------------------
// Phase 2 step s: rp = h(s-1) @ Rk + b_rec; gates; h(s).
// Fully async staging from f16 sources, double-buffered LDS:
//   issue(chunk i+1) ; s_wait_asynccnt 4 ; barrier ; WMMA(chunk i) ; barrier
// Per chunk per wave: exactly 4 async instructions (1x b128 A + 3x b64 B),
// uniform across lanes, so the ASYNCcnt discipline is exact.
// ---------------------------------------------------------------------------
__global__ __launch_bounds__(256)
void gru_step(const float* __restrict__ xp,
              const _Float16* __restrict__ rkT,     // [N3U][U_] f16
              const float* __restrict__ b_rec,
              const _Float16* __restrict__ hsrc,    // [B_][U_] f16 = h(s-1)
              _Float16* __restrict__ hdst,          // [B_][U_] f16 = h(s)
              const float* __restrict__ hprev32, int hstride,
              float* __restrict__ out, int s) {
  __shared__ __align__(16) _Float16 sA[2][B_ * LDH];
  __shared__ __align__(16) _Float16 sB[2][3][32 * LDH];

  const int t = threadIdx.x, lane = t & 31, w = t >> 5;
  const int r = w >> 1;               // row tile 0..3 (batch rows r*16..)
  const int c = w & 1;                // col tile 0..1 (units c*16..)
  const int ubase = blockIdx.x * 32;  // 16 blocks cover U=512

  auto issue = [&](int kk, int bi) {
    // A: 64 rows x 64B contiguous, b128 x4/row = 256 xfers (1/thread)
    {
      int rr = t >> 2, q = t & 3;
      async_b128(lds_off(&sA[bi][rr * LDH + q * 8]),
                 hsrc + rr * U_ + kk + q * 8);
    }
    // B: 3 gates x 32 rows x 64B, b64 x8/row = 256 xfers per gate (1/thread)
#pragma unroll
    for (int g = 0; g < 3; ++g) {
      int rr = t >> 3, q = t & 7;
      async_b64(lds_off(&sB[bi][g][rr * LDH + q * 4]),
                rkT + (g * U_ + ubase + rr) * U_ + kk + q * 4);
    }
  };

  v8f acc[3] = {};
  issue(0, 0);

  for (int it = 0; it < U_ / 32; ++it) {
    const int bi = it & 1;
    if (it + 1 < U_ / 32) {
      issue((it + 1) * 32, bi ^ 1);
      wait_async_le4();               // newest 4 outstanding = next chunk
    } else {
      wait_async0();
    }
    __syncthreads();

    v16h a = load_a_frag(&sA[bi][(r * 16 + (lane & 15)) * LDH], lane);
#pragma unroll
    for (int g = 0; g < 3; ++g) {
      v16h b = load_b_frag(&sB[bi][g][(c * 16 + (lane & 15)) * LDH], lane);
      acc[g] = __builtin_amdgcn_wmma_f32_16x16x32_f16(
          false, a, false, b, (short)0, acc[g], false, false);
    }
    __syncthreads();
  }

  // Gates: acc[0]=rz, acc[1]=rr, acc[2]=rh for this wave's 16x16 subtile.
  const int u = ubase + c * 16 + (lane & 15);
  const float bz = b_rec[u];
  const float br = b_rec[u + U_];
  const float bh = b_rec[u + 2 * U_];
#pragma unroll
  for (int v = 0; v < 8; ++v) {
    int m = r * 16 + v + ((lane >> 4) << 3);        // batch row
    long long xrow = ((long long)m * S_ + s) * N3U;
    float xz = xp[xrow + u];
    float xr = xp[xrow + U_ + u];
    float xh = xp[xrow + 2 * U_ + u];
    float h  = hprev32[m * hstride + u];
    float z  = sigmoid_f(xz + acc[0][v] + bz);
    float rr = sigmoid_f(xr + acc[1][v] + br);
    float pre = xh + rr * (acc[2][v] + bh);
    float th  = 2.0f * sigmoid_f(2.0f * pre) - 1.0f;   // tanh
    float hn  = z * h + (1.0f - z) * th;
    out[((long long)m * S_ + s) * U_ + u] = hn;        // f32 output / gate h
    hdst[m * U_ + u] = (_Float16)hn;                   // f16 feed for s+1
  }
}

// Final state = out[:, S-1, :]
__global__ __launch_bounds__(256)
void copy_state(const float* __restrict__ out, float* __restrict__ st) {
  int i = blockIdx.x * blockDim.x + threadIdx.x;
  if (i < B_ * U_) {
    int b = i / U_, u = i % U_;
    st[i] = out[((long long)b * S_ + (S_ - 1)) * U_ + u];
  }
}

// ---------------------------------------------------------------------------
// Workspace layout (bytes):
//   [0)            xp     : 8192*1536*4 = 50,331,648
//   [RKT_OFF)      rkT    : 1536*512*2  =  1,572,864
//   [KERNT_OFF)    kernT  : 1536*256*2  =    786,432
//   [HB0_OFF)      hbuf0  : 64*512*2    =     65,536
//   [HB1_OFF)      hbuf1  : 64*512*2    =     65,536   total ~50.4 MiB
// ---------------------------------------------------------------------------
#define RKT_OFF   50331648ull
#define KERNT_OFF 51904512ull
#define HB0_OFF   52690944ull
#define HB1_OFF   52756480ull

extern "C" void kernel_launch(void* const* d_in, const int* in_sizes, int n_in,
                              void* d_out, int out_size, void* d_ws, size_t ws_size,
                              hipStream_t stream) {
  (void)in_sizes; (void)n_in; (void)out_size; (void)ws_size;
  const int*   x      = (const int*)d_in[0];
  const float* hidden = (const float*)d_in[1];
  const float* emb    = (const float*)d_in[2];
  const float* kern   = (const float*)d_in[3];
  const float* rk     = (const float*)d_in[4];
  const float* bias   = (const float*)d_in[5];   // [2][1536]: b_in, b_rec
  float* out = (float*)d_out;

  char* ws = (char*)d_ws;
  float*     xp    = (float*)ws;
  _Float16*  rkT   = (_Float16*)(ws + RKT_OFF);
  _Float16*  kernT = (_Float16*)(ws + KERNT_OFF);
  _Float16*  hb[2] = { (_Float16*)(ws + HB0_OFF), (_Float16*)(ws + HB1_OFF) };

  // Setup: pre-convert stationary operands to f16 (transposed for B staging).
  transpose_cvt<<<dim3(N3U / 32, U_ / 32), 256, 0, stream>>>(rk, rkT, U_, N3U);
  transpose_cvt<<<dim3(N3U / 32, D_ / 32), 256, 0, stream>>>(kern, kernT, D_, N3U);
  cvt_f32_to_f16<<<(B_ * U_ + 255) / 256, 256, 0, stream>>>(hidden, hb[0], B_ * U_);

  // Phase 1: input projection (parallel WMMA GEMM with embedding gather).
  gru_input_proj<<<dim3(B_ * S_ / 128, N3U / 64), 256, 0, stream>>>(
      x, emb, kernT, bias, xp);

  // Phase 2: sequential recurrence; rkT (1.5 MB f16) stays L2-resident.
  const float* b_rec = bias + N3U;
  for (int s = 0; s < S_; ++s) {
    const float* hp32 = (s == 0) ? hidden : (out + (long long)(s - 1) * U_);
    int hstride       = (s == 0) ? U_ : (S_ * U_);
    gru_step<<<dim3(U_ / 32), 256, 0, stream>>>(
        xp, rkT, b_rec, hb[s & 1], hb[(s + 1) & 1], hp32, hstride, out, s);
  }

  // Final state tail of d_out.
  float* st = out + (long long)B_ * S_ * U_;
  copy_state<<<(B_ * U_ + 255) / 256, 256, 0, stream>>>(out, st);
}
